// LocalFeatureFusion_12601434046898
// MI455X (gfx1250) — matrix operand order repaired and checked by reading, not compiled, not run on hardware
//
#include <hip/hip_runtime.h>
#include <hip/hip_bf16.h>

#define DEVI __device__ __forceinline__

typedef __attribute__((ext_vector_type(16))) _Float16 v16h;
typedef __attribute__((ext_vector_type(8)))  _Float16 v8h;
typedef __attribute__((ext_vector_type(8)))  float    v8f;

// ---------------- problem constants (from reference) ----------------
constexpr int B  = 4;
constexpr int N  = 4096;
constexpr int M  = 8192;
constexpr int C  = 256;
constexpr int H  = 8;
constexpr int K  = 8;
constexpr int T  = K + 1;      // 9 tokens per window
constexpr int FF = 4 * C;      // 1024
constexpr int NW = B * N;      // 16384 windows
constexpr int WPB = 4;         // windows per block in lff_window
constexpr float R2    = 0.04f;                 // radius^2 (R=0.2)
constexpr float SCALE = 0.17677669529663687f;  // 1/sqrt(DH=32)

// ---------------- workspace layout (bytes), ~26 MB total ----------------
constexpr size_t OFF_WQ  = 0;
constexpr size_t OFF_WK  = OFF_WQ + (size_t)C * C * 2;
constexpr size_t OFF_WV  = OFF_WK + (size_t)C * C * 2;
constexpr size_t OFF_WO  = OFF_WV + (size_t)C * C * 2;
constexpr size_t OFF_W1  = OFF_WO + (size_t)C * C * 2;
constexpr size_t OFF_W2  = OFF_W1 + (size_t)C * FF * 2;
constexpr size_t OFF_IDX = OFF_W2 + (size_t)FF * C * 2;
constexpr size_t OFF_O0  = OFF_IDX + (size_t)NW * K * 4;
constexpr size_t OFF_X0  = OFF_O0 + (size_t)NW * C * 2;
// end = OFF_X0 + NW*C*4  ~= 26 MB

// ---------------- dynamic LDS partition for lff_window ----------------
constexpr int SZMAT = WPB * 16 * C * 2;          // 32768 B per f16 matrix
constexpr int LOFF_X   = 0;
constexpr int LOFF_Q   = LOFF_X + SZMAT;
constexpr int LOFF_K   = LOFF_Q + SZMAT;
constexpr int LOFF_V   = LOFF_K + SZMAT;
constexpr int LOFF_SC  = LOFF_V + SZMAT;         // WPB*H*T floats (1152 B)
constexpr int LOFF_CP  = LOFF_SC + WPB * H * T * 4 + 0;   // WPB*T*3 floats
constexpr int LOFF_NI  = LOFF_CP + WPB * T * 3 * 4;       // WPB*T ints
constexpr int LOFF_PM  = LOFF_NI + WPB * T * 4;           // WPB*T ints
constexpr int LDS_WIN  = LOFF_PM + WPB * T * 4;           // total ~132.9 KB

// ---------------- WMMA fragment helpers ----------------
// A (16x32 f16), row-major source with leading dim ld:
//   row = lane&15 ; k runs [c, c+16), c = k0 + 16*(lane>>4)
DEVI v16h load_a_frag(const _Float16* xs, int ld, int k0, int lane) {
  const int row = lane & 15;
  const int c   = k0 + ((lane >> 4) << 4);
  const _Float16* p = xs + row * ld + c;
  v8h lo = *(const v8h*)p;
  v8h hi = *(const v8h*)(p + 8);
  v16h a;
#pragma unroll
  for (int i = 0; i < 8; ++i) { a[i] = lo[i]; a[i + 8] = hi[i]; }
  return a;
}

// B (32x16 f16) from N-major (transposed) weights: wt[n*ldk + k].
//   col n = n0 + (lane&15); k runs [kc, kc+16), kc = k0 + 16*(lane>>4)
DEVI v16h load_b_frag(const _Float16* wt, int ldk, int n0, int k0, int lane) {
  const int n  = n0 + (lane & 15);
  const int kc = k0 + ((lane >> 4) << 4);
  return *(const v16h*)(wt + (size_t)n * ldk + kc);  // 32B contiguous
}

DEVI v8f wmma16(v16h a, v16h b, v8f c) {
  return __builtin_amdgcn_wmma_f32_16x16x32_f16(false, a, false, b,
                                                (short)0, c, false, false);
}

// ---------------- kernel 0: weight convert + transpose ----------------
// src: Kd x Nd row-major f32 ; dst: Nd x Kd (N-major) f16
__global__ void lff_wprep(const float* __restrict__ src,
                          _Float16* __restrict__ dst, int Kd, int Nd) {
  int i = blockIdx.x * 256 + threadIdx.x;
  if (i < Kd * Nd) {
    int n = i / Kd, k = i - n * Kd;
    dst[i] = (_Float16)src[k * Nd + n];
  }
}

// ---------------- kernel 1: radius-kNN (top-8 within R) ----------------
__global__ void __launch_bounds__(128)
lff_knn(const float* __restrict__ qxyz, const float* __restrict__ kxyz,
        const unsigned char* __restrict__ kpad, int* __restrict__ idxg) {
  __shared__ float tx[256], ty[256], tz[256];
  __shared__ unsigned char tp[256];
  const int q = blockIdx.x * 128 + threadIdx.x;  // all 128 queries share batch b
  const int b = q / N;
  const float qx = qxyz[q * 3 + 0], qy = qxyz[q * 3 + 1], qz = qxyz[q * 3 + 2];
  float bd[K]; int bi[K];
#pragma unroll
  for (int s = 0; s < K; ++s) { bd[s] = 3.0e38f; bi[s] = -1; }

  for (int t0 = 0; t0 < M; t0 += 256) {
    for (int j = threadIdx.x; j < 256; j += 128) {
      int g = b * M + t0 + j;
      tx[j] = kxyz[g * 3 + 0];
      ty[j] = kxyz[g * 3 + 1];
      tz[j] = kxyz[g * 3 + 2];
      tp[j] = kpad[g];
    }
    __syncthreads();
    for (int j = 0; j < 256; ++j) {
      float dx = qx - tx[j], dy = qy - ty[j], dz = qz - tz[j];
      float d2 = dx * dx + dy * dy + dz * dz;
      if (!tp[j] && d2 <= R2 && d2 < bd[K - 1]) {
        int p = K - 1;
        while (p > 0 && bd[p - 1] > d2) {
          bd[p] = bd[p - 1]; bi[p] = bi[p - 1]; --p;
        }
        bd[p] = d2; bi[p] = t0 + j;
      }
    }
    __syncthreads();
  }
#pragma unroll
  for (int s = 0; s < K; ++s) idxg[q * K + s] = bi[s];
}

// ---------------- kernel 2: per-window K/V/Q0 projections + attention ----------------
// 1 block = WPB(4) windows, 128 threads = 4 waves.
// Each weight B-fragment is loaded ONCE and applied to 4 windows' A-fragments
// (4 accumulators) -> 4x less L2 weight traffic than one-window-per-block.
__global__ void __launch_bounds__(128)
lff_window(const float* __restrict__ qxyz, const float* __restrict__ qfeat,
           const float* __restrict__ kxyz, const float* __restrict__ kfeat,
           const float* __restrict__ Wpos, const float* __restrict__ bpos,
           const float* __restrict__ bq, const float* __restrict__ bk,
           const float* __restrict__ bv,
           const _Float16* __restrict__ wq16, const _Float16* __restrict__ wk16,
           const _Float16* __restrict__ wv16,
           const int* __restrict__ idxg,
           _Float16* __restrict__ o0g, float* __restrict__ x0g) {
  extern __shared__ char smem[];
  _Float16* x16  = (_Float16*)(smem + LOFF_X);   // [WPB][16*C] f16
  _Float16* q16  = (_Float16*)(smem + LOFF_Q);
  _Float16* k16  = (_Float16*)(smem + LOFF_K);
  _Float16* v16  = (_Float16*)(smem + LOFF_V);
  float*    sc   = (float*)(smem + LOFF_SC);     // [WPB][H][T]
  float*    cpos = (float*)(smem + LOFF_CP);     // [WPB][T][3]
  int*      nIdx = (int*)(smem + LOFF_NI);       // [WPB][T]
  int*      padm = (int*)(smem + LOFF_PM);       // [WPB][T]

  const int w0   = blockIdx.x * WPB;
  const int tid  = threadIdx.x;
  const int lane = tid & 31;
  const int wid  = tid >> 5;

  if (tid < WPB * T) {
    int wv = tid / T, t = tid - wv * T;
    if (t == 0) { nIdx[tid] = -1; padm[tid] = 0; }
    else {
      int j = idxg[(w0 + wv) * K + t - 1];
      nIdx[tid] = j;
      padm[tid] = (j < 0) ? 1 : 0;
    }
  }
  __syncthreads();
  if (tid < WPB * T * 3) {
    int wv = tid / (T * 3), r = tid - wv * (T * 3);
    int t = r / 3, d = r - t * 3;
    int w = w0 + wv, b = w / N;
    int j = nIdx[wv * T + t];
    cpos[tid] = (j < 0) ? qxyz[(size_t)w * 3 + d]
                        : kxyz[((size_t)b * M + j) * 3 + d];
  }
  __syncthreads();

  // x = tokens + coords@Wpos + bpos  (f16 A operand; row 0 also f32->scratch)
  for (int e = tid; e < WPB * T * C; e += 128) {
    int wv = e / (T * C), r = e - wv * (T * C);
    int t = r / C, c = r - t * C;
    int w = w0 + wv, b = w / N;
    int j = nIdx[wv * T + t];
    float f = (j < 0) ? qfeat[(size_t)w * C + c]
                      : kfeat[((size_t)b * M + j) * C + c];
    const float* cp = cpos + (wv * T + t) * 3;
    float xv = f + cp[0] * Wpos[c] + cp[1] * Wpos[C + c] +
               cp[2] * Wpos[2 * C + c] + bpos[c];
    x16[wv * 16 * C + t * C + c] = (_Float16)xv;
    if (t == 0) x0g[(size_t)w * C + c] = xv;
  }
  // zero pad rows 9..15 of each window
  for (int e = tid; e < WPB * 7 * C; e += 128) {
    int wv = e / (7 * C), r = e - wv * (7 * C);
    x16[wv * 16 * C + (T + r / C) * C + (r % C)] = (_Float16)0.0f;
  }
  __syncthreads();

  // Q/K/V projections: 3 mats x 16 col-tiles = 48 wave-jobs;
  // per k-step: 1 B-fragment load, 4 WMMA (one per window).
  for (int job = wid; job < 48; job += 4) {
    const int mat = job >> 4, nt = job & 15;
    const _Float16* wt   = (mat == 0) ? wq16 : (mat == 1) ? wk16 : wv16;
    const float*    bias = (mat == 0) ? bq   : (mat == 1) ? bk   : bv;
    _Float16*       dst  = (mat == 0) ? q16  : (mat == 1) ? k16  : v16;
    v8f ac0 = {}, ac1 = {}, ac2 = {}, ac3 = {};
#pragma unroll
    for (int kk = 0; kk < C; kk += 32) {
      v16h bf = load_b_frag(wt, C, nt * 16, kk, lane);
      ac0 = wmma16(load_a_frag(x16 + 0 * 16 * C, C, kk, lane), bf, ac0);
      ac1 = wmma16(load_a_frag(x16 + 1 * 16 * C, C, kk, lane), bf, ac1);
      ac2 = wmma16(load_a_frag(x16 + 2 * 16 * C, C, kk, lane), bf, ac2);
      ac3 = wmma16(load_a_frag(x16 + 3 * 16 * C, C, kk, lane), bf, ac3);
    }
    const int col   = nt * 16 + (lane & 15);
    const float bb  = bias[col];
    const int rbase = (lane >> 4) << 3;
#pragma unroll
    for (int r = 0; r < 8; ++r) {
      int row = rbase + r;
      dst[0 * 16 * C + row * C + col] = (_Float16)(ac0[r] + bb);
      dst[1 * 16 * C + row * C + col] = (_Float16)(ac1[r] + bb);
      dst[2 * 16 * C + row * C + col] = (_Float16)(ac2[r] + bb);
      dst[3 * 16 * C + row * C + col] = (_Float16)(ac3[r] + bb);
    }
  }
  __syncthreads();

  // attention, query token 0 only: scores (WPB x 8 heads x 9 keys)
  for (int e = tid; e < WPB * H * T; e += 128) {
    int wv = e / (H * T), r = e - wv * (H * T);
    int h = r / T, tk = r - h * T;
    const _Float16* qv = q16 + wv * 16 * C + h * 32;        // row 0 of Q
    const _Float16* kv = k16 + wv * 16 * C + tk * C + h * 32;
    float s = 0.f;
#pragma unroll
    for (int d = 0; d < 32; ++d) s += (float)qv[d] * (float)kv[d];
    sc[e] = padm[wv * T + tk] ? -1e9f : s * SCALE;
  }
  __syncthreads();
  if (tid < WPB * H) {
    float* row = sc + tid * T;
    float mx = -3e38f;
    for (int tk = 0; tk < T; ++tk) mx = fmaxf(mx, row[tk]);
    float sum = 0.f;
    for (int tk = 0; tk < T; ++tk) {
      float e = __expf(row[tk] - mx);
      row[tk] = e; sum += e;
    }
    float inv = 1.f / sum;
    for (int tk = 0; tk < T; ++tk) row[tk] *= inv;
  }
  __syncthreads();
  // o0 = sum_k a[h][k] * v[k][:]  -> f16 scratch (batched GEMM input)
  for (int e = tid; e < WPB * C; e += 128) {
    int wv = e / C, c = e - wv * C;
    int h = c >> 5;
    const float* pr = sc + (wv * H + h) * T;
    float acc = 0.f;
#pragma unroll
    for (int tk = 0; tk < T; ++tk)
      acc += pr[tk] * (float)v16[wv * 16 * C + tk * C + c];
    o0g[(size_t)(w0 + wv) * C + c] = (_Float16)acc;
  }
}

// ---------------- kernel 3: batched row-0 GEMMs (Wo, FFN) + LN + skips ----------------
// 1 block = 32 windows, 256 threads = 8 waves, 128KB dynamic LDS.
// Each B-fragment loaded once, applied to both 16-row M-tiles (2 accumulators).
__global__ void __launch_bounds__(256)
lff_rows(const _Float16* __restrict__ o0g, const float* __restrict__ x0g,
         const float* __restrict__ qfeat,
         const _Float16* __restrict__ wo16, const float* __restrict__ bo,
         const float* __restrict__ g1, const float* __restrict__ be1,
         const _Float16* __restrict__ w116, const float* __restrict__ b1,
         const _Float16* __restrict__ w216, const float* __restrict__ b2,
         const float* __restrict__ g2, const float* __restrict__ be2,
         float* __restrict__ out) {
  extern __shared__ char smem[];
  _Float16* a16  = (_Float16*)smem;                                   // 16 KB
  _Float16* h16  = (_Float16*)(smem + 32 * C * 2);                    // 64 KB
  float*    buf  = (float*)(smem + 32 * C * 2 + 32 * FF * 2);         // 32 KB
  _Float16* x16b = (_Float16*)(smem + 32 * C * 2 + 32 * FF * 2 + 32 * C * 4); // 16 KB

  const int w0   = blockIdx.x * 32;
  const int tid  = threadIdx.x;
  const int lane = tid & 31;
  const int wid  = tid >> 5;

  for (int e = tid; e < 32 * C; e += 256) a16[e] = o0g[(size_t)w0 * C + e];
  __syncthreads();

  // GEMM1: attn = o0 @ WoT ; buf = attn + bo + x0 (residual)
  for (int nt = wid; nt < 16; nt += 8) {
    v8f ac0 = {}, ac1 = {};
#pragma unroll
    for (int kk = 0; kk < C; kk += 32) {
      v16h bf = load_b_frag(wo16, C, nt * 16, kk, lane);
      ac0 = wmma16(load_a_frag(a16, C, kk, lane), bf, ac0);
      ac1 = wmma16(load_a_frag(a16 + 16 * C, C, kk, lane), bf, ac1);
    }
    int col = nt * 16 + (lane & 15);
    float bb = bo[col];
    int rb = (lane >> 4) << 3;
#pragma unroll
    for (int r = 0; r < 8; ++r) {
      int r0 = rb + r, r1 = 16 + rb + r;
      buf[r0 * C + col] = ac0[r] + bb + x0g[(size_t)(w0 + r0) * C + col];
      buf[r1 * C + col] = ac1[r] + bb + x0g[(size_t)(w0 + r1) * C + col];
    }
  }
  __syncthreads();

  // LN1 (per row): buf <- x1 (f32), x16b <- x1 (f16)
  if (tid < 32) {
    float mu = 0.f;
    for (int c = 0; c < C; ++c) mu += buf[tid * C + c];
    mu *= (1.f / C);
    float var = 0.f;
    for (int c = 0; c < C; ++c) { float d = buf[tid * C + c] - mu; var += d * d; }
    var *= (1.f / C);
    float rs = rsqrtf(var + 1e-5f);
    for (int c = 0; c < C; ++c) {
      float xv = (buf[tid * C + c] - mu) * rs * g1[c] + be1[c];
      buf[tid * C + c]  = xv;
      x16b[tid * C + c] = (_Float16)xv;
    }
  }
  __syncthreads();

  // GEMM2: h = relu(x1 @ W1T + b1)  (32 x 1024, f16)
  for (int nt = wid; nt < 64; nt += 8) {
    v8f ac0 = {}, ac1 = {};
#pragma unroll
    for (int kk = 0; kk < C; kk += 32) {
      v16h bf = load_b_frag(w116, C, nt * 16, kk, lane);
      ac0 = wmma16(load_a_frag(x16b, C, kk, lane), bf, ac0);
      ac1 = wmma16(load_a_frag(x16b + 16 * C, C, kk, lane), bf, ac1);
    }
    int col = nt * 16 + (lane & 15);
    float bb = b1[col];
    int rb = (lane >> 4) << 3;
#pragma unroll
    for (int r = 0; r < 8; ++r) {
      h16[(rb + r) * FF + col]      = (_Float16)fmaxf(ac0[r] + bb, 0.f);
      h16[(16 + rb + r) * FF + col] = (_Float16)fmaxf(ac1[r] + bb, 0.f);
    }
  }
  __syncthreads();

  // GEMM3: y = h @ W2T + b2 + x1 (residual, in-place into buf)
  for (int nt = wid; nt < 16; nt += 8) {
    v8f ac0 = {}, ac1 = {};
    for (int kk = 0; kk < FF; kk += 32) {
      __builtin_prefetch((const void*)(w216 + (size_t)(nt * 16 + (lane & 15)) * FF + kk + 64), 0, 1);
      v16h bf = load_b_frag(w216, FF, nt * 16, kk, lane);
      ac0 = wmma16(load_a_frag(h16, FF, kk, lane), bf, ac0);
      ac1 = wmma16(load_a_frag(h16 + 16 * FF, FF, kk, lane), bf, ac1);
    }
    int col = nt * 16 + (lane & 15);
    float bb = b2[col];
    int rb = (lane >> 4) << 3;
#pragma unroll
    for (int r = 0; r < 8; ++r) {
      int r0 = rb + r, r1 = 16 + rb + r;
      buf[r0 * C + col] = ac0[r] + bb + buf[r0 * C + col];
      buf[r1 * C + col] = ac1[r] + bb + buf[r1 * C + col];
    }
  }
  __syncthreads();

  // LN2 + final skip (q_feat) -> out
  if (tid < 32) {
    float mu = 0.f;
    for (int c = 0; c < C; ++c) mu += buf[tid * C + c];
    mu *= (1.f / C);
    float var = 0.f;
    for (int c = 0; c < C; ++c) { float d = buf[tid * C + c] - mu; var += d * d; }
    var *= (1.f / C);
    float rs = rsqrtf(var + 1e-5f);
    size_t base = (size_t)(w0 + tid) * C;
    for (int c = 0; c < C; ++c) {
      float xv = (buf[tid * C + c] - mu) * rs * g2[c] + be2[c];
      out[base + c] = xv + qfeat[base + c];
    }
  }
}

// ---------------- launcher ----------------
extern "C" void kernel_launch(void* const* d_in, const int* in_sizes, int n_in,
                              void* d_out, int out_size, void* d_ws, size_t ws_size,
                              hipStream_t stream) {
  (void)in_sizes; (void)n_in; (void)out_size; (void)ws_size;

  const float* qxyz  = (const float*)d_in[0];
  const float* qfeat = (const float*)d_in[1];
  const float* kxyz  = (const float*)d_in[2];
  const float* kfeat = (const float*)d_in[3];
  const unsigned char* kpad = (const unsigned char*)d_in[4];
  const float* Wpos = (const float*)d_in[5];
  const float* bpos = (const float*)d_in[6];
  const float* Wq = (const float*)d_in[7];
  const float* bq = (const float*)d_in[8];
  const float* Wk = (const float*)d_in[9];
  const float* bk = (const float*)d_in[10];
  const float* Wv = (const float*)d_in[11];
  const float* bv = (const float*)d_in[12];
  const float* Wo = (const float*)d_in[13];
  const float* bo = (const float*)d_in[14];
  const float* g1  = (const float*)d_in[15];
  const float* be1 = (const float*)d_in[16];
  const float* W1 = (const float*)d_in[17];
  const float* b1 = (const float*)d_in[18];
  const float* W2 = (const float*)d_in[19];
  const float* b2 = (const float*)d_in[20];
  const float* g2  = (const float*)d_in[21];
  const float* be2 = (const float*)d_in[22];
  float* out = (float*)d_out;

  char* ws = (char*)d_ws;
  _Float16* wq16 = (_Float16*)(ws + OFF_WQ);
  _Float16* wk16 = (_Float16*)(ws + OFF_WK);
  _Float16* wv16 = (_Float16*)(ws + OFF_WV);
  _Float16* wo16 = (_Float16*)(ws + OFF_WO);
  _Float16* w116 = (_Float16*)(ws + OFF_W1);
  _Float16* w216 = (_Float16*)(ws + OFF_W2);
  int*      idxg = (int*)(ws + OFF_IDX);
  _Float16* o0g  = (_Float16*)(ws + OFF_O0);
  float*    x0g  = (float*)(ws + OFF_X0);

  // weight prep: f32 -> f16, transposed to N-major
  lff_wprep<<<(C * C + 255) / 256, 256, 0, stream>>>(Wq, wq16, C, C);
  lff_wprep<<<(C * C + 255) / 256, 256, 0, stream>>>(Wk, wk16, C, C);
  lff_wprep<<<(C * C + 255) / 256, 256, 0, stream>>>(Wv, wv16, C, C);
  lff_wprep<<<(C * C + 255) / 256, 256, 0, stream>>>(Wo, wo16, C, C);
  lff_wprep<<<(C * FF + 255) / 256, 256, 0, stream>>>(W1, w116, C, FF);   // Kd=C, Nd=FF
  lff_wprep<<<(FF * C + 255) / 256, 256, 0, stream>>>(W2, w216, FF, C);   // Kd=FF, Nd=C

  // radius-kNN
  lff_knn<<<NW / 128, 128, 0, stream>>>(qxyz, kxyz, kpad, idxg);

  // per-window projections + attention (q0 only), 4 windows/block
  lff_window<<<NW / WPB, 128, LDS_WIN, stream>>>(qxyz, qfeat, kxyz, kfeat,
                                                 Wpos, bpos, bq, bk, bv,
                                                 wq16, wk16, wv16,
                                                 idxg, o0g, x0g);

  // batched row-0 GEMMs + LN + skips
  lff_rows<<<NW / 32, 256, 131072, stream>>>(o0g, x0g, qfeat, wo16, bo,
                                             g1, be1, w116, b1, w216, b2,
                                             g2, be2, out);
}
